// Mamba2D_40020505264687
// MI455X (gfx1250) — matrix operand order repaired and checked by reading, not compiled, not run on hardware
//
#include <hip/hip_runtime.h>
#include <hip/hip_bf16.h>

// ---------------------------------------------------------------------------
// Mamba-2D bidirectional selective-SSM block for gfx1250 (MI455X, wave32).
// GEMMs run on v_wmma_f32_16x16x32_f16 with 2x2 register blocking (32x32 per
// wave, 2 b128 loads per WMMA); scan is register-resident VALU.
// ---------------------------------------------------------------------------

typedef __attribute__((ext_vector_type(8)))  _Float16 v8h;
typedef __attribute__((ext_vector_type(16))) _Float16 v16h;
typedef __attribute__((ext_vector_type(8)))  float    v8f;

#define BATCH   4
#define SEQ_L   2048
#define DM      512
#define DI      1024
#define DS      16
#define TOKENS  (BATCH * SEQ_L)       // 8192 rows per direction
#define NDIR    4

// ---------------- f32 -> f16 convert -------------------------------------
__global__ void cvt_f16_kernel(const float* __restrict__ src,
                               _Float16* __restrict__ dst, long n) {
    long i = (long)blockIdx.x * blockDim.x + threadIdx.x;
    if (i < n) dst[i] = (_Float16)src[i];
}

// ---------------- transpose [R,C] -> [C,R] with f16 convert ---------------
__global__ void transpose_cvt_kernel(const float* __restrict__ src,
                                     _Float16* __restrict__ dst, int R, int C) {
    long n = (long)R * C;
    long i = (long)blockIdx.x * blockDim.x + threadIdx.x;
    if (i >= n) return;
    long base = (long)blockIdx.y * n;
    int r = (int)(i / C), c = (int)(i % C);
    dst[base + (long)c * R + r] = (_Float16)src[base + i];
}

// ---------------- fragment loaders (wave32 WMMA layouts) -------------------
// A frag (16x32 f16): halves 0..7 = K[base..+7], 8..15 = K[base+16..+7]
__device__ __forceinline__ v16h load_frag_a(const _Float16* p) {
    v8h x0 = *(const v8h*)p;
    v8h x1 = *(const v8h*)(p + 16);
    return __builtin_shufflevector(x0, x1, 0,1,2,3,4,5,6,7,8,9,10,11,12,13,14,15);
}
// B frag (32x16 f16): halves 0..7 = K[base..+7], 8..15 = K[base+16..+7]
__device__ __forceinline__ v16h load_frag_b(const _Float16* p) {
    v8h x0 = *(const v8h*)p;
    v8h x1 = *(const v8h*)(p + 16);
    return __builtin_shufflevector(x0, x1, 0,1,2,3,4,5,6,7,8,9,10,11,12,13,14,15);
}

// ---------------- generic WMMA GEMM: C[M,N] = A[M,K] * Bt[N,K]^T ----------
// One wave computes a 32x32 tile (2x2 of 16x16). lda == ldb == K, ldc == N.
// flipMask: per-dir bit -> A rows are read sequence-reversed within a batch.
__global__ void wmma_gemm_tn(const _Float16* __restrict__ A,
                             const _Float16* __restrict__ Bt,
                             float* __restrict__ C,
                             int M, int N, int K,
                             long strideA, long strideB, long strideC,
                             int flipMask, int L) {
    const int dir = blockIdx.y;
    const _Float16* Ad = A + (size_t)dir * strideA;
    const _Float16* Bd = Bt + (size_t)dir * strideB;
    float*           Cd = C + (size_t)dir * strideC;

    const int lane = threadIdx.x & 31;
    const int wave = threadIdx.x >> 5;
    const int tilesN = N >> 5;
    const long tile = (long)blockIdx.x * (blockDim.x >> 5) + wave;
    const long tilesTotal = (long)(M >> 5) * tilesN;
    if (tile >= tilesTotal) return;               // wave-uniform: EXEC stays full

    const int tm = (int)(tile / tilesN);
    const int tn = (int)(tile % tilesN);
    const int hi = (lane >> 4) & 1;
    const bool flip = (flipMask >> dir) & 1;      // dir-uniform

    int ar0 = tm * 32 + (lane & 15);
    int ar1 = ar0 + 16;
    if (flip) {
        int l0 = ar0 % L; ar0 = ar0 - l0 + (L - 1 - l0);
        int l1 = ar1 % L; ar1 = ar1 - l1 + (L - 1 - l1);
    }
    const int br0 = tn * 32 + (lane & 15);
    const int br1 = br0 + 16;

    const _Float16* ap0 = Ad + (size_t)ar0 * K + hi * 8;
    const _Float16* ap1 = Ad + (size_t)ar1 * K + hi * 8;
    const _Float16* bp0 = Bd + (size_t)br0 * K + hi * 8;
    const _Float16* bp1 = Bd + (size_t)br1 * K + hi * 8;

    v8f acc00 = {}, acc01 = {}, acc10 = {}, acc11 = {};
    for (int k0 = 0; k0 < K; k0 += 32) {
        v16h a0 = load_frag_a(ap0 + k0);
        v16h a1 = load_frag_a(ap1 + k0);
        v16h b0 = load_frag_b(bp0 + k0);
        v16h b1 = load_frag_b(bp1 + k0);
        acc00 = __builtin_amdgcn_wmma_f32_16x16x32_f16(false, a0, false, b0,
                                                       (short)0, acc00, false, false);
        acc01 = __builtin_amdgcn_wmma_f32_16x16x32_f16(false, a0, false, b1,
                                                       (short)0, acc01, false, false);
        acc10 = __builtin_amdgcn_wmma_f32_16x16x32_f16(false, a1, false, b0,
                                                       (short)0, acc10, false, false);
        acc11 = __builtin_amdgcn_wmma_f32_16x16x32_f16(false, a1, false, b1,
                                                       (short)0, acc11, false, false);
    }

    // C/D layout: VGPR r -> M = r + hi*8, lane&15 -> N
    v8f accs[2][2] = {{acc00, acc01}, {acc10, acc11}};
#pragma unroll
    for (int i = 0; i < 2; ++i) {
#pragma unroll
        for (int j = 0; j < 2; ++j) {
            const int cm0 = tm * 32 + i * 16 + hi * 8;
            const int cn  = tn * 32 + j * 16 + (lane & 15);
#pragma unroll
            for (int r = 0; r < 8; ++r)
                Cd[(size_t)(cm0 + r) * N + cn] = accs[i][j][r];
        }
    }
}

// ---------------- causal depthwise conv (k=4) + bias + silu ---------------
__global__ void conv_silu_kernel(const float* __restrict__ xz,
                                 const float* __restrict__ conv_w,
                                 const float* __restrict__ conv_b,
                                 float* __restrict__ xc) {
    const size_t total = (size_t)NDIR * TOKENS * DI;
    size_t i = (size_t)blockIdx.x * blockDim.x + threadIdx.x;
    if (i >= total) return;
    int c = (int)(i & (DI - 1));
    size_t tok = i >> 10;                 // dir*8192 + b*L + l
    int dir = (int)(tok >> 13);
    int l = (int)(tok & (SEQ_L - 1));
    const float* w = conv_w + ((size_t)dir * DI + c) * 4;
    float acc = conv_b[dir * DI + c];
#pragma unroll
    for (int k = 0; k < 4; ++k) {
        int l2 = l - 3 + k;
        if (l2 >= 0)
            acc += w[k] * xz[(tok + (size_t)(l2 - l)) * (2 * DI) + c];
    }
    acc = acc / (1.f + __expf(-acc));     // silu
    xc[i] = acc;
}

// ---------------- x_proj: [1024] -> delta(1)+B(16)+C(16), softplus(delta) --
__global__ void xproj_kernel(const float* __restrict__ xc,
                             const float* __restrict__ xproj_w,
                             float* __restrict__ dbc) {
    size_t tok = blockIdx.x;              // dir*8192 + row
    int dir = (int)(tok >> 13);
    __shared__ float sx[DI];
    for (int i = threadIdx.x; i < DI; i += 64) sx[i] = xc[tok * DI + i];
    __syncthreads();
    int j = threadIdx.x;
    if (j < 33) {
        const float* w = xproj_w + (size_t)dir * DI * 33;
        float acc = 0.f;
        for (int k = 0; k < DI; ++k) acc += sx[k] * w[k * 33 + j];
        if (j == 0) acc = (acc > 20.f) ? acc : log1pf(__expf(acc));
        dbc[tok * 33 + j] = acc;
    }
}

// ---------------- selective scan: h[16] in registers per channel -----------
// grid: 64 blocks = dir(4) x batch(4) x channel-group(4); 256 threads.
// Fuses +xc*D, silu(z) gate, f16 convert for out_proj.
__global__ void scan_kernel(const float* __restrict__ dbc,
                            const float* __restrict__ xc,
                            const float* __restrict__ xz,
                            const float* __restrict__ A_log,
                            const float* __restrict__ D_param,
                            _Float16* __restrict__ y16) {
    int bx = blockIdx.x;
    int dir = bx >> 4, b = (bx >> 2) & 3, grp = bx & 3;
    int c = grp * 256 + threadIdx.x;

    float A[DS], h[DS];
#pragma unroll
    for (int s = 0; s < DS; ++s) {
        A[s] = -__expf(A_log[((size_t)dir * DI + c) * DS + s]);
        h[s] = 0.f;
    }
    float Dp = D_param[dir * DI + c];

    __shared__ float sh[33];
    for (int l = 0; l < SEQ_L; ++l) {
        size_t row = (size_t)dir * TOKENS + (size_t)b * SEQ_L + l;
        if (threadIdx.x < 33) sh[threadIdx.x] = dbc[row * 33 + threadIdx.x];
        __syncthreads();
        float delta = sh[0];
        float xval  = xc[row * DI + c];
        float y = 0.f;
#pragma unroll
        for (int s = 0; s < DS; ++s) {
            float dA = __expf(delta * A[s]);
            h[s] = dA * h[s] + (delta * sh[1 + s]) * xval;
            y += h[s] * sh[17 + s];
        }
        y += xval * Dp;
        float z = xz[row * (2 * DI) + DI + c];
        y *= z / (1.f + __expf(-z));      // * silu(z)
        y16[row * DI + c] = (_Float16)y;
        __syncthreads();
    }
}

// ---------------- combine fwd/bwd halves, build f16 gate input -------------
__global__ void combine_kernel(const float* __restrict__ outs,
                               float* __restrict__ h_h, float* __restrict__ h_v,
                               _Float16* __restrict__ hcat) {
    size_t i = (size_t)blockIdx.x * blockDim.x + threadIdx.x;
    if (i >= (size_t)TOKENS * DM) return;
    int c = (int)(i & (DM - 1));
    size_t row = i >> 9;
    int b = (int)(row / SEQ_L), l = (int)(row % SEQ_L);
    size_t brow = (size_t)b * SEQ_L + (SEQ_L - 1 - l);
    const size_t DSZ = (size_t)TOKENS * DM;
    float hh = 0.5f * (outs[0 * DSZ + row * DM + c] + outs[1 * DSZ + brow * DM + c]);
    float hv = 0.5f * (outs[2 * DSZ + row * DM + c] + outs[3 * DSZ + brow * DM + c]);
    h_h[row * DM + c] = hh;
    h_v[row * DM + c] = hv;
    hcat[row * (2 * DM) + c]      = (_Float16)hh;
    hcat[row * (2 * DM) + DM + c] = (_Float16)hv;
}

// ---------------- sigmoid gate + residual + layernorm ----------------------
__global__ void final_kernel(const float* __restrict__ glin,
                             const float* __restrict__ gate_b,
                             const float* __restrict__ h_h,
                             const float* __restrict__ h_v,
                             const float* __restrict__ x,
                             const float* __restrict__ ln_w,
                             const float* __restrict__ ln_b,
                             float* __restrict__ out) {
    size_t row = blockIdx.x;              // 0..8191
    __shared__ float ssum[256], ssq[256];
    float yv[2], lsum = 0.f, lsq = 0.f;
#pragma unroll
    for (int i = 0; i < 2; ++i) {
        int c = threadIdx.x + i * 256;
        float g = 1.f / (1.f + __expf(-(glin[row * DM + c] + gate_b[c])));
        float h = g * h_h[row * DM + c] + (1.f - g) * h_v[row * DM + c];
        float y = h + x[row * DM + c];
        yv[i] = y; lsum += y; lsq += y * y;
    }
    ssum[threadIdx.x] = lsum; ssq[threadIdx.x] = lsq;
    __syncthreads();
    for (int off = 128; off > 0; off >>= 1) {
        if (threadIdx.x < off) {
            ssum[threadIdx.x] += ssum[threadIdx.x + off];
            ssq[threadIdx.x]  += ssq[threadIdx.x + off];
        }
        __syncthreads();
    }
    float mu  = ssum[0] * (1.f / DM);
    float var = ssq[0] * (1.f / DM) - mu * mu;
    float inv = rsqrtf(var + 1e-5f);
#pragma unroll
    for (int i = 0; i < 2; ++i) {
        int c = threadIdx.x + i * 256;
        out[row * DM + c] = (yv[i] - mu) * inv * ln_w[c] + ln_b[c];
    }
}

// ---------------------------------------------------------------------------
extern "C" void kernel_launch(void* const* d_in, const int* in_sizes, int n_in,
                              void* d_out, int out_size, void* d_ws, size_t ws_size,
                              hipStream_t stream) {
    const float* x       = (const float*)d_in[0];
    const float* in_w    = (const float*)d_in[1];
    const float* conv_w  = (const float*)d_in[2];
    const float* conv_b  = (const float*)d_in[3];
    const float* xproj_w = (const float*)d_in[4];
    const float* A_log   = (const float*)d_in[5];
    const float* D_param = (const float*)d_in[6];
    const float* out_w   = (const float*)d_in[7];
    const float* gate_w  = (const float*)d_in[8];
    const float* gate_b  = (const float*)d_in[9];
    const float* ln_w    = (const float*)d_in[10];
    const float* ln_b    = (const float*)d_in[11];
    float* out = (float*)d_out;

    char* ws = (char*)d_ws;
    size_t off = 0;
    auto alloc = [&](size_t bytes) {
        size_t o = off;
        off += (bytes + 255) & ~(size_t)255;
        return o;
    };
    float*    xz     = (float*)   (ws + alloc((size_t)NDIR * TOKENS * 2 * DI * 4));
    float*    xc     = (float*)   (ws + alloc((size_t)NDIR * TOKENS * DI * 4));
    float*    dbc    = (float*)   (ws + alloc((size_t)NDIR * TOKENS * 33 * 4));
    _Float16* y16    = (_Float16*)(ws + alloc((size_t)NDIR * TOKENS * DI * 2));
    float*    outs   = (float*)   (ws + alloc((size_t)NDIR * TOKENS * DM * 4));
    float*    h_h    = (float*)   (ws + alloc((size_t)TOKENS * DM * 4));
    float*    h_v    = (float*)   (ws + alloc((size_t)TOKENS * DM * 4));
    _Float16* hcat   = (_Float16*)(ws + alloc((size_t)TOKENS * 2 * DM * 2));
    float*    glin   = (float*)   (ws + alloc((size_t)TOKENS * DM * 4));
    _Float16* x16    = (_Float16*)(ws + alloc((size_t)TOKENS * DM * 2));
    _Float16* inwt   = (_Float16*)(ws + alloc((size_t)NDIR * 2 * DI * DM * 2));
    _Float16* outwt  = (_Float16*)(ws + alloc((size_t)NDIR * DM * DI * 2));
    _Float16* gatewt = (_Float16*)(ws + alloc((size_t)DM * 2 * DM * 2));

    // 1) precision conversion + weight transposes (f16, [N,K] layout)
    {
        long n = (long)TOKENS * DM;
        cvt_f16_kernel<<<(n + 255) / 256, 256, 0, stream>>>(x, x16, n);
        transpose_cvt_kernel<<<dim3((DM * 2 * DI + 255) / 256, NDIR), 256, 0, stream>>>(
            in_w, inwt, DM, 2 * DI);
        transpose_cvt_kernel<<<dim3((DI * DM + 255) / 256, NDIR), 256, 0, stream>>>(
            out_w, outwt, DI, DM);
        transpose_cvt_kernel<<<dim3((2 * DM * DM + 255) / 256, 1), 256, 0, stream>>>(
            gate_w, gatewt, 2 * DM, DM);
    }

    // 2) in_proj GEMM: xz[dir] = xin[dir] @ in_w[dir]   (flip dirs 1,3 via indexing)
    {
        long tiles = (long)(TOKENS / 32) * (2 * DI / 32);   // 256*64
        wmma_gemm_tn<<<dim3((unsigned)((tiles + 7) / 8), NDIR), 256, 0, stream>>>(
            x16, inwt, xz, TOKENS, 2 * DI, DM,
            /*strideA*/ 0, /*strideB*/ (long)2 * DI * DM, /*strideC*/ (long)TOKENS * 2 * DI,
            /*flipMask*/ 0xA, SEQ_L);
    }

    // 3) causal depthwise conv + silu
    {
        size_t n = (size_t)NDIR * TOKENS * DI;
        conv_silu_kernel<<<(unsigned)((n + 255) / 256), 256, 0, stream>>>(
            xz, conv_w, conv_b, xc);
    }

    // 4) x_proj -> delta/B/C (softplus on delta)
    xproj_kernel<<<NDIR * TOKENS, 64, 0, stream>>>(xc, xproj_w, dbc);

    // 5) selective scan (fused D-skip, silu(z) gate, f16 convert)
    scan_kernel<<<64, 256, 0, stream>>>(dbc, xc, xz, A_log, D_param, y16);

    // 6) out_proj GEMM: outs[dir] = y[dir] @ out_w[dir]
    {
        long tiles = (long)(TOKENS / 32) * (DM / 32);       // 256*16
        wmma_gemm_tn<<<dim3((unsigned)((tiles + 7) / 8), NDIR), 256, 0, stream>>>(
            y16, outwt, outs, TOKENS, DM, DI,
            (long)TOKENS * DI, (long)DM * DI, (long)TOKENS * DM, 0, SEQ_L);
    }

    // 7) combine fwd/bwd, build gate input
    {
        size_t n = (size_t)TOKENS * DM;
        combine_kernel<<<(unsigned)((n + 255) / 256), 256, 0, stream>>>(
            outs, h_h, h_v, hcat);
    }

    // 8) gate GEMM: glin = [h_h|h_v] @ gate_w
    {
        long tiles = (long)(TOKENS / 32) * (DM / 32);
        wmma_gemm_tn<<<dim3((unsigned)((tiles + 7) / 8), 1), 256, 0, stream>>>(
            hcat, gatewt, glin, TOKENS, DM, 2 * DM, 0, 0, 0, 0, SEQ_L);
    }

    // 9) sigmoid gate + residual + layernorm
    final_kernel<<<TOKENS, 256, 0, stream>>>(glin, gate_b, h_h, h_v, x, ln_w, ln_b, out);
}
